// QLSTM_65481071400953
// MI455X (gfx1250) — compile-verified, matrix operand order checked by reading, not run on hardware
//
#include <hip/hip_runtime.h>

// ---------------------------------------------------------------------------
// QLSTM forward for MI455X (gfx1250), persistent-kernel version.
//   z = [x_t, h] @ Wcomb + b4 ; Wcomb rows 0..D-1 = Wx, rows D.. = Wh + R.
//   - bf16 WMMA (v_wmma_f32_16x16x32_bf16), fp32 accumulate
//   - per-WG weight slice resident in LDS (128 KB of 320 KB) for all 512 steps
//   - cell state c resident in registers for the whole sequence
//   - h kept in WMMA A-fragment layout (double buffered) -> b128 loads
//   - steps separated by a device-scope atomic grid barrier
// ---------------------------------------------------------------------------

typedef __attribute__((ext_vector_type(16))) __bf16 v16bf;
typedef __attribute__((ext_vector_type(8)))  float  v8f;

#define QT 512
#define QB 64
#define QD 512
#define QH 512
#define NWG 32          // one WG per 16-wide j slice
#define STACKED (QT * QB * QH)

static __device__ __forceinline__ unsigned short f32_to_bf16_bits(float f) {
  union { float f; unsigned u; } cv; cv.f = f;
  unsigned r = cv.u + 0x7FFFu + ((cv.u >> 16) & 1u); // round-to-nearest-even
  return (unsigned short)(r >> 16);
}
static __device__ __forceinline__ float sigmoidf_(float x) {
  return 1.0f / (1.0f + __expf(-x));
}
static __device__ __forceinline__ v8f wmma_bf16(v16bf a, v16bf b, v8f c) {
  return __builtin_amdgcn_wmma_f32_16x16x32_bf16(
      false, a, false, b, (short)0, c, false, false);
}
// 16-bit A/B fragment: lane-group g (=lane>>4), element e (0..15)
static __device__ __forceinline__ int klocal(int g, int e) {
  return (e < 8) ? (g * 8 + e) : (16 + g * 8 + (e - 8));
}

// ---------------------------------------------------------------------------
// Prep: Wfrag [kb 0..31][nt 0..127][lane 0..31][e 0..15] bf16
// ---------------------------------------------------------------------------
__global__ __launch_bounds__(256) void qlstm_prep_wfrag(
    const float* __restrict__ Wf, const float* __restrict__ Wi,
    const float* __restrict__ Wg, const float* __restrict__ Wo,
    const float* __restrict__ Rf, const float* __restrict__ Ri,
    const float* __restrict__ Rg, const float* __restrict__ Ro,
    unsigned short* __restrict__ wfrag) {
  int idx = blockIdx.x * 256 + threadIdx.x;
  if (idx >= 32 * 128 * 32 * 16) return;
  int e    = idx & 15;
  int lane = (idx >> 4) & 31;
  int blk  = idx >> 9;
  int nt   = blk & 127;
  int kb   = blk >> 7;
  int n    = nt * 16 + (lane & 15);
  int k    = kb * 32 + klocal(lane >> 4, e);
  int G    = n >> 9;
  int j    = n & 511;
  const float* W4[4] = {Wf, Wi, Wg, Wo};
  const float* R4[4] = {Rf, Ri, Rg, Ro};
  float val = W4[G][k * QH + j];
  if (k >= QD) val += R4[G][(k - QD) * QH + j];
  wfrag[idx] = f32_to_bf16_bits(val);
}

// ---------------------------------------------------------------------------
// Prep: Xfrag [t][kb 0..15][mi 0..3][lane 0..31][e 0..15] bf16 (A fragments)
// ---------------------------------------------------------------------------
__global__ __launch_bounds__(256) void qlstm_prep_xfrag(
    const float* __restrict__ x, unsigned short* __restrict__ xfrag) {
  int idx = blockIdx.x * 256 + threadIdx.x;
  if (idx >= QT * QB * QD) return;
  int e    = idx & 15;
  int lane = (idx >> 4) & 31;
  int r    = idx >> 9;
  int mi   = r & 3;
  int r2   = r >> 2;
  int kb   = r2 & 15;
  int t    = r2 >> 4;
  int m    = (lane & 15) + mi * 16;
  int k    = kb * 32 + klocal(lane >> 4, e);
  xfrag[idx] = f32_to_bf16_bits(x[(t * QB + m) * QD + k]);
}

__global__ __launch_bounds__(256) void qlstm_init_state(
    unsigned short* __restrict__ hA, unsigned short* __restrict__ hB,
    unsigned* __restrict__ bar) {
  int i = blockIdx.x * 256 + threadIdx.x;
  if (i < QB * QH) { hA[i] = 0; hB[i] = 0; }
  if (i == 0) *bar = 0u;
}

// ---------------------------------------------------------------------------
// Persistent recurrence kernel. 32 WGs x 128 threads (4 waves, 1 per SIMD32).
//   WG jb owns columns j in [jb*16, jb*16+16) of every gate.
//   Wave mi owns batch rows [mi*16, mi*16+16).
// ---------------------------------------------------------------------------
__global__ __launch_bounds__(128) void qlstm_persist(
    const unsigned short* __restrict__ xfrag,   // [T][32768] bf16 A-fragments
    const unsigned short* __restrict__ wfrag,   // 2M bf16 B-fragments
    unsigned short* __restrict__ hfragA,        // [32768] bf16 A-fragments
    unsigned short* __restrict__ hfragB,
    unsigned* __restrict__ bar,
    const float* __restrict__ bF, const float* __restrict__ bI,
    const float* __restrict__ bG, const float* __restrict__ bO,
    float* __restrict__ out)
{
  // this WG's weight slice: [kb 0..31][G 0..3][lane][e] = 128 KB, LDS-resident
  __shared__ __align__(16) unsigned short wtile[32 * 4 * 32 * 16];

  const int tid  = threadIdx.x;
  const int lane = tid & 31;
  const int mi   = tid >> 5;          // wave id = M tile
  const int jb   = blockIdx.x;        // 0..31

  // ---- one-time stage of weights into LDS (coalesced b128 copies) ----
  for (int ci = tid; ci < 4096; ci += 128) {       // 4096 chunks of 32 B
    int lc = ci & 31, G = (ci >> 5) & 3, kb = ci >> 7;
    const uint4* src = (const uint4*)(wfrag +
        (((size_t)(kb * 128 + G * 32 + jb)) << 9) + lc * 16);
    uint4* dst = (uint4*)(wtile + ((kb * 4 + G) << 9) + lc * 16);
    dst[0] = src[0];
    dst[1] = src[1];
  }
  __syncthreads();

  const int g   = lane >> 4;
  const int np  = lane & 15;
  const int j   = jb * 16 + np;
  const int gp8 = g * 8;
  const float bfv = bF[j], biv = bI[j], bgv = bG[j], bov = bO[j];

  // h-fragment scatter-store coordinates for this lane's produced (b, j)
  const int kbw = jb >> 1;                          // k-block of column j
  const int gw  = np >> 3;                          // consumer lane group
  const int ew  = (np & 7) + ((jb & 1) << 3);       // consumer element

  float creg[8];                                    // cell state, in registers
#pragma unroll
  for (int r = 0; r < 8; ++r) creg[r] = 0.0f;

  for (int t = 0; t < QT; ++t) {
    const unsigned short* hr = (t & 1) ? hfragB : hfragA;
    unsigned short*       hw = (t & 1) ? hfragA : hfragB;
    const unsigned short* xf = xfrag + (size_t)t * (QB * QD);

    v8f accF = {}, accI = {}, accG = {}, accO = {};

    // ---- x contribution: K 0..511, A from global Xfrag, B from LDS ----
    for (int kb = 0; kb < 16; ++kb) {
      v16bf a = *(const v16bf*)(xf + (((kb * 4 + mi) * 32 + lane) << 4));
      const unsigned short* wl = wtile + ((kb * 4) << 9) + (lane << 4);
      v16bf b0 = *(const v16bf*)(wl + (0 << 9));
      v16bf b1 = *(const v16bf*)(wl + (1 << 9));
      v16bf b2 = *(const v16bf*)(wl + (2 << 9));
      v16bf b3 = *(const v16bf*)(wl + (3 << 9));
      accF = wmma_bf16(a, b0, accF);
      accI = wmma_bf16(a, b1, accI);
      accG = wmma_bf16(a, b2, accG);
      accO = wmma_bf16(a, b3, accO);
    }
    // ---- h contribution: K 512..1023, A from h fragments, B from LDS ----
    for (int kb = 0; kb < 16; ++kb) {
      v16bf a = *(const v16bf*)(hr + (((kb * 4 + mi) * 32 + lane) << 4));
      const unsigned short* wl = wtile + (((kb + 16) * 4) << 9) + (lane << 4);
      v16bf b0 = *(const v16bf*)(wl + (0 << 9));
      v16bf b1 = *(const v16bf*)(wl + (1 << 9));
      v16bf b2 = *(const v16bf*)(wl + (2 << 9));
      v16bf b3 = *(const v16bf*)(wl + (3 << 9));
      accF = wmma_bf16(a, b0, accF);
      accI = wmma_bf16(a, b1, accI);
      accG = wmma_bf16(a, b2, accG);
      accO = wmma_bf16(a, b3, accO);
    }

    // ---- prefetch next timestep's A fragments while finishing this step ----
    if (t + 1 < QT) {
      const unsigned short* xn = xf + (QB * QD);
#pragma unroll
      for (int kb = 0; kb < 16; ++kb)
        __builtin_prefetch(xn + (((kb * 4 + mi) * 32 + lane) << 4), 0, 1);
    }

    // ---- lane-local gate math + state update ----
    float* out_t = out + (size_t)t * (QB * QH);
#pragma unroll
    for (int r = 0; r < 8; ++r) {
      int brow = mi * 16 + gp8 + r;     // C/D layout: lanes 16-31 hold M+8
      float fz = sigmoidf_(accF[r] + bfv);
      float iz = sigmoidf_(accI[r] + biv);
      float gz = tanhf(accG[r] + bgv);
      float oz = sigmoidf_(accO[r] + bov);
      float cn = fz * creg[r] + iz * gz;
      creg[r]  = cn;
      float hn = oz * tanhf(cn);
      out_t[brow * QH + j] = hn;
      // scatter h into next step's A-fragment slot
      hw[(((kbw * 4 + mi) * 32) + gw * 16 + gp8 + r) * 16 + ew] =
          f32_to_bf16_bits(hn);
      if (t == QT - 1) out[STACKED + brow * QH + j] = hn;      // hx
    }

    // ---- grid barrier: release h writes, acquire everyone else's ----
    __syncthreads();
    if (tid == 0) {
      __hip_atomic_fetch_add(bar, 1u, __ATOMIC_RELEASE,
                             __HIP_MEMORY_SCOPE_AGENT);
      unsigned tgt = (unsigned)(t + 1) * (unsigned)NWG;
      while (__hip_atomic_load(bar, __ATOMIC_ACQUIRE,
                               __HIP_MEMORY_SCOPE_AGENT) < tgt)
        __builtin_amdgcn_s_sleep(1);
    }
    __syncthreads();
  }

  // ---- final cell state from registers ----
#pragma unroll
  for (int r = 0; r < 8; ++r)
    out[STACKED + QB * QH + (mi * 16 + gp8 + r) * QH + j] = creg[r];
}

// ---------------------------------------------------------------------------
extern "C" void kernel_launch(void* const* d_in, const int* in_sizes, int n_in,
                              void* d_out, int out_size, void* d_ws,
                              size_t ws_size, hipStream_t stream) {
  const float* x  = (const float*)d_in[0];
  const float* Wf = (const float*)d_in[1];
  const float* bf = (const float*)d_in[2];
  const float* Wi = (const float*)d_in[3];
  const float* bi = (const float*)d_in[4];
  const float* Wg = (const float*)d_in[5];
  const float* bg = (const float*)d_in[6];
  const float* Wo = (const float*)d_in[7];
  const float* bo = (const float*)d_in[8];
  const float* Rf = (const float*)d_in[9];
  const float* Ri = (const float*)d_in[10];
  const float* Rg = (const float*)d_in[11];
  const float* Ro = (const float*)d_in[12];
  float* out = (float*)d_out;

  // workspace: Wfrag 4MB | Xfrag 32MB | hfragA 64KB | hfragB 64KB | bar
  unsigned short* wfrag  = (unsigned short*)d_ws;              // 2,097,152 el
  unsigned short* xfrag  = wfrag + (size_t)2097152;            // 16,777,216 el
  unsigned short* hfragA = xfrag + (size_t)16777216;           // 32,768 el
  unsigned short* hfragB = hfragA + 32768;                     // 32,768 el
  unsigned*       bar    = (unsigned*)(hfragB + 32768);

  qlstm_prep_wfrag<<<2097152 / 256, 256, 0, stream>>>(Wf, Wi, Wg, Wo,
                                                      Rf, Ri, Rg, Ro, wfrag);
  qlstm_prep_xfrag<<<16777216 / 256, 256, 0, stream>>>(x, xfrag);
  qlstm_init_state<<<(QB * QH + 255) / 256, 256, 0, stream>>>(hfragA, hfragB,
                                                              bar);
  qlstm_persist<<<NWG, 128, 0, stream>>>(xfrag, wfrag, hfragA, hfragB, bar,
                                         bf, bi, bg, bo, out);
}
